// MultiHeadAttention_9062380994866
// MI455X (gfx1250) — matrix operand order repaired and checked by reading, not compiled, not run on hardware
//
#include <hip/hip_runtime.h>
#include <hip/hip_bf16.h>

// ---------------- problem constants ----------------
#define E_ 1024
#define H_ 16
#define D_ 64
#define B_ 2
#define S_ 2048
#define M_ (B_*S_)   // 4096 rows of token matrix

typedef unsigned short u16;
typedef unsigned int   u32;

typedef __attribute__((ext_vector_type(16))) __bf16 bf16x16;
typedef __attribute__((ext_vector_type(8)))  float  f32x8;

// One WMMA 16x16x32 bf16 operand = 16 bf16 per lane = 32 bytes = 2x uint4
union FragBF { bf16x16 v; uint4 q[2]; };

__device__ __forceinline__ u16 f2bf(float f) {
  u32 u = __float_as_uint(f);
  u32 r = u + 0x7FFFu + ((u >> 16) & 1u);   // round-to-nearest-even
  return (u16)(r >> 16);
}

__device__ __forceinline__ f32x8 wmma_bf16(const FragBF& a, const FragBF& b, f32x8 c) {
  return __builtin_amdgcn_wmma_f32_16x16x32_bf16(false, a.v, false, b.v,
                                                 (short)0, c, false, false);
}

// Generic LDS pointer low 32 bits == wave-relative LDS byte address (ISA §10.2:
// LDS aperture generic address = {SHARED_BASE, addr[31:0]}).
__device__ __forceinline__ unsigned lds_off32(const void* p) {
  return (unsigned)(unsigned long long)p;
}

// Async DMA: global -> LDS, 16 bytes per lane, tracked by ASYNCcnt.
__device__ __forceinline__ void async_b128(unsigned lds_byte, const void* gaddr) {
  asm volatile("global_load_async_to_lds_b128 %0, %1, off"
               :: "v"(lds_byte), "v"(gaddr) : "memory");
}

// ---------------- conversion kernels ----------------
__global__ void cvt_f32_to_bf16(const float* __restrict__ in, u16* __restrict__ out, int n) {
  int i = blockIdx.x * blockDim.x + threadIdx.x;
  if (i < n) out[i] = f2bf(in[i]);
}

// Wt[n*E + k] = bf16(W[k*E + n])  (transpose so B-fragment columns are contiguous in k)
__global__ void cvt_w_transpose(const float* __restrict__ W, u16* __restrict__ Wt) {
  int i = blockIdx.x * blockDim.x + threadIdx.x;   // over E*E
  int n = i >> 10, k = i & (E_ - 1);
  Wt[i] = f2bf(W[k * E_ + n]);
}

// ---------------- bf16 WMMA GEMM ----------------
// C[m][n] = sum_k A[m][k] * Wt[n][k] + bias[n]
// A: [M_,E_] bf16 row-major, Wt: [E_,E_] bf16 row-major (transposed weight)
// mode 0: out u16 bf16, per-head Q/K layout  [(b*H+h)*S + s]*D + d
// mode 1: out u16 bf16, per-head V-transposed layout [(b*H+h)*D + d]*S + s
// mode 2: out f32 row-major [M_,E_]  (final output)
__global__ __launch_bounds__(256)
void gemm_bf16(const u16* __restrict__ A, const u16* __restrict__ Wt,
               const float* __restrict__ bias, void* __restrict__ outp, int mode) {
  const int lane = threadIdx.x & 31;
  const int wid  = threadIdx.x >> 5;     // 8 waves
  const int l16  = lane & 15;
  const int hl   = lane >> 4;
  const int waveM = wid >> 1;            // 0..3 -> 32 rows each
  const int waveN = wid & 1;             // 0..1 -> 64 cols each
  const int m_base = blockIdx.x * 128 + waveM * 32;
  const int n_base = blockIdx.y * 128 + waveN * 64;

  f32x8 acc[2][4];
  #pragma unroll
  for (int mt = 0; mt < 2; ++mt)
    #pragma unroll
    for (int nt = 0; nt < 4; ++nt)
      #pragma unroll
      for (int r = 0; r < 8; ++r) acc[mt][nt][r] = 0.0f;

  for (int k0 = 0; k0 < E_; k0 += 32) {
    FragBF a[2];
    #pragma unroll
    for (int mt = 0; mt < 2; ++mt) {
      // A layout: lane half hl holds K = k0+8*hl+(0..7) and k0+16+8*hl+(0..7)
      const u16* p = A + (size_t)(m_base + mt * 16 + l16) * E_ + k0 + 8 * hl;
      a[mt].q[0] = *(const uint4*)p;
      a[mt].q[1] = *(const uint4*)(p + 16);
    }
    #pragma unroll
    for (int nt = 0; nt < 4; ++nt) {
      // B layout: lane half hl holds K = k0+8*hl+(0..7) and k0+16+8*hl+(0..7), column n per lane
      FragBF bfr;
      const u16* p = Wt + (size_t)(n_base + nt * 16 + l16) * E_ + k0 + 8 * hl;
      bfr.q[0] = *(const uint4*)p;
      bfr.q[1] = *(const uint4*)(p + 16);
      #pragma unroll
      for (int mt = 0; mt < 2; ++mt)
        acc[mt][nt] = wmma_bf16(a[mt], bfr, acc[mt][nt]);
    }
  }

  #pragma unroll
  for (int mt = 0; mt < 2; ++mt) {
    #pragma unroll
    for (int nt = 0; nt < 4; ++nt) {
      const int n = n_base + nt * 16 + l16;
      const float bv = bias[n];
      #pragma unroll
      for (int r = 0; r < 8; ++r) {
        const int m = m_base + mt * 16 + r + 8 * hl;  // C layout: VGPR r -> row r+8*half
        const float val = acc[mt][nt][r] + bv;
        if (mode == 2) {
          ((float*)outp)[(size_t)m * E_ + n] = val;
        } else {
          const int b = m >> 11, s = m & (S_ - 1);
          const int h = n >> 6,  d = n & (D_ - 1);
          const size_t idx = (mode == 0)
              ? ((size_t)(b * H_ + h) * S_ + s) * D_ + d
              : ((size_t)(b * H_ + h) * D_ + d) * S_ + s;
          ((u16*)outp)[idx] = f2bf(val);
        }
      }
    }
  }
}

// ---------------- flash attention (per (b,h), online softmax) ----------------
// Q,K: [B*H][S][D] bf16 ;  Vt: [B*H][D][S] bf16 ;  mask: [B][S] u8 (nonzero = masked)
// Out: Ob [B][S][E] bf16 (heads merged)
// K/V key-blocks (64 keys) are staged into LDS once per workgroup with
// GLOBAL_LOAD_ASYNC_TO_LDS (double-buffered, ASYNCcnt-synchronized) instead of
// each of the 8 waves redundantly loading the same fragments from L2.
#define NKB (S_/64)
#define KROW 72            // padded row stride (u16 elems), 144B = 16B-aligned

__global__ __launch_bounds__(256)
void flash_attn(const u16* __restrict__ Qb, const u16* __restrict__ Kb,
                const u16* __restrict__ Vtb, const unsigned char* __restrict__ maskp,
                u16* __restrict__ Ob) {
  __shared__ u16 Kt[2][64][KROW];     // K block: [key][d]
  __shared__ u16 Vt[2][64][KROW];     // V block: [d][key]
  __shared__ u16 Plds[8][16][KROW];   // per-wave P transit (C-layout -> A-layout)

  const int lane = threadIdx.x & 31;
  const int wid  = threadIdx.x >> 5;
  const int l16  = lane & 15;
  const int hl   = lane >> 4;
  const int bh = blockIdx.y;
  const int b  = bh >> 4;            // / H
  const int hh = bh & (H_ - 1);
  const int q0 = blockIdx.x * 128 + wid * 16;   // 16 query rows per wave

  const unsigned kt_base = lds_off32(&Kt[0][0][0]);
  const unsigned vt_base = lds_off32(&Vt[0][0][0]);
  const unsigned buf_stride = 64 * KROW * 2;    // bytes per buffer

  // Q fragments (kept resident; K-dim = D = 64 -> two 32-deep fragments)
  FragBF qf[2];
  {
    const u16* p = Qb + ((size_t)bh * S_ + q0 + l16) * D_;
    #pragma unroll
    for (int kk = 0; kk < 2; ++kk) {
      qf[kk].q[0] = *(const uint4*)(p + 32 * kk + 8 * hl);
      qf[kk].q[1] = *(const uint4*)(p + 32 * kk + 16 + 8 * hl);
    }
  }

  float mrow[8], lrow[8];
  f32x8 oacc[4];
  #pragma unroll
  for (int r = 0; r < 8; ++r) { mrow[r] = -1e30f; lrow[r] = 0.0f; }
  #pragma unroll
  for (int dt = 0; dt < 4; ++dt)
    #pragma unroll
    for (int r = 0; r < 8; ++r) oacc[dt][r] = 0.0f;

  // Issue this workgroup's share (4 b128/wave) of async K+V tile loads for block kb.
  auto issue_tile = [&](int kb, int buf) {
    const u16* kg = Kb + ((size_t)bh * S_ + (size_t)kb * 64) * D_;  // 8KB contiguous
    #pragma unroll
    for (int t = 0; t < 2; ++t) {
      const int i = wid * 32 + lane + t * 256;    // 16B-chunk id, 0..511
      const int row = i >> 3, c = i & 7;
      async_b128(kt_base + (unsigned)buf * buf_stride + row * (KROW*2) + c * 16,
                 kg + (size_t)i * 8);
    }
    #pragma unroll
    for (int t = 0; t < 2; ++t) {
      const int i = wid * 32 + lane + t * 256;
      const int d = i >> 3, c = i & 7;            // V rows: 128B each, stride S_
      async_b128(vt_base + (unsigned)buf * buf_stride + d * (KROW*2) + c * 16,
                 Vtb + ((size_t)bh * D_ + d) * S_ + (size_t)kb * 64 + c * 8);
    }
  };

  const float scale = 0.125f;  // 1/sqrt(D)

  issue_tile(0, 0);

  for (int kb = 0; kb < NKB; ++kb) {
    const int buf = kb & 1;
    if (kb + 1 < NKB) {
      issue_tile(kb + 1, (kb + 1) & 1);
      asm volatile("s_wait_asynccnt 0x4" ::: "memory");  // tile kb done (in-order)
    } else {
      asm volatile("s_wait_asynccnt 0x0" ::: "memory");
    }
    __syncthreads();   // all waves' async writes visible

    // ---- scores S = (Q K^T) * scale, key-padding mask ----
    f32x8 sc[4];
    bool cm[4];
    #pragma unroll
    for (int nt = 0; nt < 4; ++nt) {
      const int kl = nt * 16 + l16;                    // key within block (column per lane)
      f32x8 z;
      #pragma unroll
      for (int r = 0; r < 8; ++r) z[r] = 0.0f;
      FragBF kf;
      kf.q[0] = *(const uint4*)&Kt[buf][kl][8 * hl];
      kf.q[1] = *(const uint4*)&Kt[buf][kl][16 + 8 * hl];
      z = wmma_bf16(qf[0], kf, z);
      kf.q[0] = *(const uint4*)&Kt[buf][kl][32 + 8 * hl];
      kf.q[1] = *(const uint4*)&Kt[buf][kl][32 + 16 + 8 * hl];
      z = wmma_bf16(qf[1], kf, z);
      const bool mk = maskp[b * S_ + kb * 64 + kl] != 0;
      cm[nt] = mk;
      #pragma unroll
      for (int r = 0; r < 8; ++r) sc[nt][r] = mk ? -1e30f : z[r] * scale;
    }

    // ---- online softmax (row = r + 8*hl lives in one 16-lane half) ----
    #pragma unroll
    for (int r = 0; r < 8; ++r) {
      float v = fmaxf(fmaxf(sc[0][r], sc[1][r]), fmaxf(sc[2][r], sc[3][r]));
      v = fmaxf(v, __shfl_xor(v, 1));
      v = fmaxf(v, __shfl_xor(v, 2));
      v = fmaxf(v, __shfl_xor(v, 4));
      v = fmaxf(v, __shfl_xor(v, 8));
      const float m_new = fmaxf(mrow[r], v);
      const float c = __expf(mrow[r] - m_new);
      float rsum = 0.0f;
      #pragma unroll
      for (int nt = 0; nt < 4; ++nt) {
        const float pp = cm[nt] ? 0.0f : __expf(sc[nt][r] - m_new);  // masked cols -> exact 0
        Plds[wid][r + 8 * hl][nt * 16 + l16] = f2bf(pp);
        rsum += pp;
      }
      rsum += __shfl_xor(rsum, 1);
      rsum += __shfl_xor(rsum, 2);
      rsum += __shfl_xor(rsum, 4);
      rsum += __shfl_xor(rsum, 8);
      lrow[r] = lrow[r] * c + rsum;
      mrow[r] = m_new;
      #pragma unroll
      for (int dt = 0; dt < 4; ++dt) oacc[dt][r] *= c;
    }
    __syncthreads();

    // ---- O += P @ V ----
    #pragma unroll
    for (int kk2 = 0; kk2 < 2; ++kk2) {
      FragBF pf;
      pf.q[0] = *(const uint4*)&Plds[wid][l16][32 * kk2 + 8 * hl];
      pf.q[1] = *(const uint4*)&Plds[wid][l16][32 * kk2 + 16 + 8 * hl];
      #pragma unroll
      for (int dt = 0; dt < 4; ++dt) {
        const int d = dt * 16 + l16;
        FragBF vf;
        vf.q[0] = *(const uint4*)&Vt[buf][d][32 * kk2 + 8 * hl];
        vf.q[1] = *(const uint4*)&Vt[buf][d][32 * kk2 + 16 + 8 * hl];
        oacc[dt] = wmma_bf16(pf, vf, oacc[dt]);
      }
    }
    __syncthreads();   // buffer free before next issue overwrites it
  }

  // ---- normalize + store merged-head bf16 output ----
  #pragma unroll
  for (int dt = 0; dt < 4; ++dt) {
    #pragma unroll
    for (int r = 0; r < 8; ++r) {
      const int s = q0 + r + 8 * hl;
      const float l = lrow[r];
      const float val = (l > 0.0f) ? (oacc[dt][r] / l) : 0.0f;  // fully-masked rows -> 0
      Ob[((size_t)(b * S_) + s) * E_ + hh * D_ + dt * 16 + l16] = f2bf(val);
    }
  }
}

// ---------------- host orchestration ----------------
extern "C" void kernel_launch(void* const* d_in, const int* in_sizes, int n_in,
                              void* d_out, int out_size, void* d_ws, size_t ws_size,
                              hipStream_t stream) {
  (void)in_sizes; (void)n_in; (void)out_size; (void)ws_size;
  const float* x  = (const float*)d_in[0];
  const unsigned char* mask = (const unsigned char*)d_in[1];  // jnp bool_ = 1 byte
  const float* Wq = (const float*)d_in[2];
  const float* bq = (const float*)d_in[3];
  const float* Wk = (const float*)d_in[4];
  const float* bk = (const float*)d_in[5];
  const float* Wv = (const float*)d_in[6];
  const float* bv = (const float*)d_in[7];
  const float* Wo = (const float*)d_in[8];
  const float* bo = (const float*)d_in[9];
  float* out = (float*)d_out;

  // workspace layout (bf16): 4*4M (X,Q,K,Vt) + 4M (O) + 4*1M (weights) = 24M elems = 48 MB
  u16* Xb  = (u16*)d_ws;
  u16* Wtq = Xb  + (size_t)M_ * E_;
  u16* Wtk = Wtq + (size_t)E_ * E_;
  u16* Wtv = Wtk + (size_t)E_ * E_;
  u16* Wto = Wtv + (size_t)E_ * E_;
  u16* Qb  = Wto + (size_t)E_ * E_;
  u16* Kb  = Qb  + (size_t)M_ * E_;
  u16* Vtb = Kb  + (size_t)M_ * E_;
  u16* Ob  = Vtb + (size_t)M_ * E_;

  const int nx = M_ * E_;
  cvt_f32_to_bf16<<<(nx + 255) / 256, 256, 0, stream>>>(x, Xb, nx);
  const int nw = E_ * E_;
  cvt_w_transpose<<<(nw + 255) / 256, 256, 0, stream>>>(Wq, Wtq);
  cvt_w_transpose<<<(nw + 255) / 256, 256, 0, stream>>>(Wk, Wtk);
  cvt_w_transpose<<<(nw + 255) / 256, 256, 0, stream>>>(Wv, Wtv);
  cvt_w_transpose<<<(nw + 255) / 256, 256, 0, stream>>>(Wo, Wto);

  dim3 g(M_ / 128, E_ / 128);
  gemm_bf16<<<g, 256, 0, stream>>>(Xb, Wtq, bq, (void*)Qb, 0);
  gemm_bf16<<<g, 256, 0, stream>>>(Xb, Wtk, bk, (void*)Kb, 0);
  gemm_bf16<<<g, 256, 0, stream>>>(Xb, Wtv, bv, (void*)Vtb, 1);

  flash_attn<<<dim3(S_ / 128, B_ * H_), 256, 0, stream>>>(Qb, Kb, Vtb, mask, Ob);

  gemm_bf16<<<g, 256, 0, stream>>>(Ob, Wto, bo, (void*)out, 2);
}
